// GRUCell_83760452207457
// MI455X (gfx1250) — matrix operand order repaired and checked by reading, not compiled, not run on hardware
//
#include <hip/hip_runtime.h>
#include <hip/hip_bf16.h>

#define GRU_BATCH 16384
#define GRU_K     512            // IN_DIM + UNITS
#define GRU_N     256            // UNITS

// tiling
#define MT        64             // rows per block
#define THREADS   512            // 16 wave32s
#define A_STRIDE  520            // padded bf16 row stride for concat tile (1040 B)
#define RP_STRIDE 264            // padded bf16 row stride for r*prev_h tile (528 B)

// shared memory partition (bf16 element offsets)
#define A_ELEMS   (MT * A_STRIDE)            // 33280
#define B_OFF     A_ELEMS                    // B region: 4096 x 16B chunks
#define B_CHUNKS  4096                       //  (2 bufs x 2048 chunks)
#define RP_OFF    (B_OFF + B_CHUNKS * 8)     // 66048
#define SM_ELEMS  (RP_OFF + MT * RP_STRIDE)  // 82944  -> 165888 bytes
#define B_BYTE    (B_OFF * 2)                // 66560: LDS byte offset of B region

// packed-weight workspace layout (16B chunks):
//  ZR region: 16384 chunks = [kt(16)][g(2)][nt(16)][h(2)][lane(32)] x 8 bf16
//  S  region:  8192 chunks = [kt(16)][nt(16)][h(2)][lane(32)] x 8 bf16
#define ZR_CHUNKS 16384
#define S_CHUNKS  8192

typedef __attribute__((ext_vector_type(16))) __bf16 v16bf;
typedef __attribute__((ext_vector_type(8)))  __bf16 v8bf;
typedef __attribute__((ext_vector_type(8)))  float  v8f;
typedef __attribute__((ext_vector_type(4)))  unsigned int v4u;
typedef __attribute__((ext_vector_type(8)))  int  v8i;
typedef __attribute__((ext_vector_type(4)))  int  v4i;

static __device__ __forceinline__ __bf16 f2bf(float f) {
    unsigned u = __builtin_bit_cast(unsigned, f);
    u += 0x7FFFu + ((u >> 16) & 1u);              // round-to-nearest-even
    unsigned short h = (unsigned short)(u >> 16);
    return __builtin_bit_cast(__bf16, h);
}

static __device__ __forceinline__ v8f wmma_bf16(v16bf a, v16bf b, v8f c) {
    return __builtin_amdgcn_wmma_f32_16x16x32_bf16(
        false, a, false, b, (short)0, c, false, false);
}

union AV { v16bf v; v8bf h[2]; };

static __device__ __forceinline__ float sigmoidf_(float x) {
    return 1.0f / (1.0f + __expf(-x));
}

// ---- TDM: DMA a contiguous block (n8 x 8 bytes) from global into LDS ----
// D# per CDNA5 ISA ch.8: group0 = {count, lds_addr, global_addr, type=2},
// group1 = {data_size=8B, tensor_dim0=tile_dim0=n8, dim1=1, stride0=n8}.
static __device__ __forceinline__ void tdm_load_slab(const void* gsrc,
                                                     unsigned lds_byte,
                                                     unsigned n8) {
    size_t ga = (size_t)gsrc;
    v4u g0;
    g0.x = 1u;                                            // count=1, user desc
    g0.y = lds_byte;                                      // lds_addr (bytes)
    g0.z = (unsigned)(ga & 0xFFFFFFFFu);                  // global_addr[31:0]
    g0.w = (unsigned)((ga >> 32) & 0x01FFFFFFu)           // global_addr[56:32]
           | 0x80000000u;                                 // type = 2 ("image")
    v8i g1;
    g1[0] = (int)(3u << 16);                              // data_size = 3 (8 B)
    g1[1] = (int)(n8 << 16);                              // tensor_dim0[15:0]
    g1[2] = (int)((n8 >> 16) | (1u << 16));               // dim0[31:16], tensor_dim1=1
    g1[3] = (int)(n8 << 16);                              // tile_dim0
    g1[4] = 0;                                            // tile_dim1=0, tile_dim2=0
    g1[5] = (int)n8;                                      // tensor_dim0_stride[31:0]
    g1[6] = 0;
    g1[7] = 0;
    v4i z4 = {0, 0, 0, 0};                                // groups 2/3: <=2D tensor
    v8i z8 = {0, 0, 0, 0, 0, 0, 0, 0};
    __builtin_amdgcn_tensor_load_to_lds(g0, g1, z4, z4, z8, 0);
}

// ---- one-shot weight pack: fp32 -> bf16, swizzled into WMMA operand chunks ----
__global__ __launch_bounds__(256) void pack_weights(
    const float* __restrict__ Wz, const float* __restrict__ Wr,
    const float* __restrict__ Ws, v8bf* __restrict__ wsp)
{
    int t = blockIdx.x * 256 + threadIdx.x;       // 0 .. 24575
    const float* W;
    int kt, l, h, nt;
    size_t dstc;
    if (t < ZR_CHUNKS) {
        int z = t;
        l = z & 31; h = (z >> 5) & 1; nt = (z >> 6) & 15;
        int g = (z >> 10) & 1; kt = z >> 11;
        W = g ? Wr : Wz;
        dstc = (size_t)z;
    } else {
        int s = t - ZR_CHUNKS;
        l = s & 31; h = (s >> 5) & 1; nt = (s >> 6) & 15; kt = s >> 10;
        W = Ws;
        dstc = (size_t)ZR_CHUNKS + s;
    }
    const float* src = W + (size_t)(kt * 32 + h * 16 + (l >> 4) * 8) * GRU_N + nt * 16 + (l & 15);
    union { v8bf v; __bf16 e[8]; } o;
    o.e[0] = f2bf(src[0 * GRU_N]); o.e[1] = f2bf(src[1 * GRU_N]); o.e[2] = f2bf(src[2 * GRU_N]); o.e[3] = f2bf(src[3 * GRU_N]);
    o.e[4] = f2bf(src[4 * GRU_N]); o.e[5] = f2bf(src[5 * GRU_N]); o.e[6] = f2bf(src[6 * GRU_N]); o.e[7] = f2bf(src[7 * GRU_N]);
    wsp[dstc] = o.v;
}

__global__ __launch_bounds__(THREADS) void gru_cell_wmma(
    const float* __restrict__ x, const float* __restrict__ prev_h,
    const v8bf* __restrict__ Wzr, const v8bf* __restrict__ Wsc,
    const float* __restrict__ bz, const float* __restrict__ br,
    const float* __restrict__ bs, float* __restrict__ out)
{
    extern __shared__ __align__(16) __bf16 sm[];
    __bf16* lds_a  = sm;                          // [64][520] concat tile
    v8bf*   lds_b  = (v8bf*)(sm + B_OFF);         // weight slabs, double buffered
    __bf16* lds_rp = sm + RP_OFF;                 // [64][264] r*prev_h

    const int m0   = blockIdx.x * MT;
    const int tid  = threadIdx.x;
    const int wave = tid >> 5;
    const int lane = tid & 31;
    const int mi   = wave >> 2;                   // M-tile 0..3
    const int ntg  = (wave & 3) * 4;              // first of 4 N-tiles
    const int r15  = lane & 15;
    const int kb   = (lane >> 4) << 3;            // A K phase 0/8; also C/D row off
    const int hi8  = kb;

    // ---- kick TDM for ZR slab kt=0 (overlaps the A-tile conversion below) ----
    if (wave == 0)
        tdm_load_slab(Wzr, B_BYTE, 4096);         // 32 KB -> buf 0

    // ---- stage concat tile [prev_h | x] (64 x 512) as bf16, padded rows ----
#pragma unroll 8
    for (int i = 0; i < 64; ++i) {
        int idx = i * THREADS + tid;              // 0..32767, coalesced
        int row = idx >> 9;
        int k   = idx & 511;
        float v = (k < GRU_N)
                    ? prev_h[(size_t)(m0 + row) * GRU_N + k]
                    : x[(size_t)(m0 + row) * GRU_N + (k - GRU_N)];
        lds_a[row * A_STRIDE + k] = f2bf(v);
    }
    if (wave == 0)
        __builtin_amdgcn_s_wait_tensorcnt(0);
    __syncthreads();

    // ---- Phase 1: z and r GEMMs over K = 512, TDM double-buffered B ----
    v8f accz[4] = {v8f{}, v8f{}, v8f{}, v8f{}};
    v8f accr[4] = {v8f{}, v8f{}, v8f{}, v8f{}};
    for (int kt = 0; kt < 16; ++kt) {
        const v8bf* bcur = lds_b + (kt & 1) * 2048;
        if (wave == 0 && kt < 15)                 // DMA next 32 KB slab
            tdm_load_slab(Wzr + (size_t)(kt + 1) * 2048,
                          B_BYTE + (unsigned)(((kt & 1) ^ 1) * 32768), 4096);
        AV a;
        const __bf16* ap = lds_a + (mi * 16 + r15) * A_STRIDE + kt * 32 + kb;
        a.h[0] = *(const v8bf*)(ap);
        a.h[1] = *(const v8bf*)(ap + 16);
#pragma unroll
        for (int j = 0; j < 4; ++j) {
            int nt = ntg + j;
            AV b0, b1;
            b0.h[0] = bcur[(nt * 2 + 0) * 32 + lane];          // gate z
            b0.h[1] = bcur[(nt * 2 + 1) * 32 + lane];
            b1.h[0] = bcur[1024 + (nt * 2 + 0) * 32 + lane];   // gate r
            b1.h[1] = bcur[1024 + (nt * 2 + 1) * 32 + lane];
            accz[j] = wmma_bf16(a.v, b0.v, accz[j]);
            accr[j] = wmma_bf16(a.v, b1.v, accr[j]);
        }
        if (wave == 0 && kt < 15)
            __builtin_amdgcn_s_wait_tensorcnt(0);
        __syncthreads();
    }

    // ---- kick TDM for S slab kt=0 (phase-1 reads of lds_b are complete) ----
    if (wave == 0)
        tdm_load_slab(Wsc, B_BYTE, 2048);         // 16 KB -> buf 0

    // ---- epilogue 1: z = sigmoid, rp = sigmoid(r)*prev_h -> LDS ----
    float zv[4][8], ph[4][8];
#pragma unroll
    for (int j = 0; j < 4; ++j) {
        int col = (ntg + j) * 16 + r15;
        float bzv = bz[col], brv = br[col];
#pragma unroll
        for (int jj = 0; jj < 8; ++jj) {
            int lrow = mi * 16 + hi8 + jj;        // C/D layout row
            float p  = prev_h[(size_t)(m0 + lrow) * GRU_N + col];
            ph[j][jj] = p;
            zv[j][jj] = sigmoidf_(accz[j][jj] + bzv);
            float rr  = sigmoidf_(accr[j][jj] + brv);
            lds_rp[lrow * RP_STRIDE + col] = f2bf(rr * p);
        }
    }
    if (wave == 0)
        __builtin_amdgcn_s_wait_tensorcnt(0);
    __syncthreads();

    // ---- Phase 2: candidate GEMM [r*prev_h | x] @ Ws ----
    v8f accs[4] = {v8f{}, v8f{}, v8f{}, v8f{}};
    for (int kt = 0; kt < 16; ++kt) {
        const v8bf* bcur = lds_b + (kt & 1) * 1024;
        if (wave == 0 && kt < 15)                 // DMA next 16 KB slab
            tdm_load_slab(Wsc + (size_t)(kt + 1) * 1024,
                          B_BYTE + (unsigned)(((kt & 1) ^ 1) * 16384), 2048);
        AV a;
        if (kt < 8) {                             // k < 256: r*prev_h from LDS
            const __bf16* ap = lds_rp + (mi * 16 + r15) * RP_STRIDE + kt * 32 + kb;
            a.h[0] = *(const v8bf*)(ap);
            a.h[1] = *(const v8bf*)(ap + 16);
        } else {                                  // k >= 256: x part of concat
            const __bf16* ap = lds_a + (mi * 16 + r15) * A_STRIDE + kt * 32 + kb;
            a.h[0] = *(const v8bf*)(ap);
            a.h[1] = *(const v8bf*)(ap + 16);
        }
#pragma unroll
        for (int j = 0; j < 4; ++j) {
            int nt = ntg + j;
            AV b;
            b.h[0] = bcur[(nt * 2 + 0) * 32 + lane];
            b.h[1] = bcur[(nt * 2 + 1) * 32 + lane];
            accs[j] = wmma_bf16(a.v, b.v, accs[j]);
        }
        if (wave == 0 && kt < 15)
            __builtin_amdgcn_s_wait_tensorcnt(0);
        __syncthreads();
    }

    // ---- epilogue 2: h_new = (1-z)*prev_h + z*tanh(cand + bs) ----
#pragma unroll
    for (int j = 0; j < 4; ++j) {
        int col = (ntg + j) * 16 + r15;
        float bsv = bs[col];
#pragma unroll
        for (int jj = 0; jj < 8; ++jj) {
            int lrow = mi * 16 + hi8 + jj;
            float ht = tanhf(accs[j][jj] + bsv);
            out[(size_t)(m0 + lrow) * GRU_N + col] =
                (1.0f - zv[j][jj]) * ph[j][jj] + zv[j][jj] * ht;
        }
    }
}

extern "C" void kernel_launch(void* const* d_in, const int* in_sizes, int n_in,
                              void* d_out, int out_size, void* d_ws, size_t ws_size,
                              hipStream_t stream) {
    (void)in_sizes; (void)n_in; (void)ws_size; (void)out_size;
    const float* x      = (const float*)d_in[0];
    const float* prev_h = (const float*)d_in[1];
    const float* Wz     = (const float*)d_in[2];
    const float* bz     = (const float*)d_in[3];
    const float* Wr     = (const float*)d_in[4];
    const float* br     = (const float*)d_in[5];
    const float* Ws     = (const float*)d_in[6];
    const float* bs     = (const float*)d_in[7];
    float* out = (float*)d_out;

    v8bf* Wpk = (v8bf*)d_ws;                      // 24576 x 16B chunks = 384 KB
    const v8bf* Wzr = Wpk;                        // ZR region: 16384 chunks
    const v8bf* Wsc = Wpk + ZR_CHUNKS;            // S region:   8192 chunks

    // pack + swizzle fp32 weights -> bf16 operand chunks (stream-ordered)
    dim3 pgrid((ZR_CHUNKS + S_CHUNKS) / 256);     // 96 blocks
    pack_weights<<<pgrid, 256, 0, stream>>>(Wz, Wr, Ws, Wpk);

    dim3 grid(GRU_BATCH / MT);                    // 256 blocks: 64 rows x 256 cols
    size_t smem = (size_t)SM_ELEMS * sizeof(__bf16);  // 165888 B
    gru_cell_wmma<<<grid, THREADS, smem, stream>>>(x, prev_h, Wzr, Wsc, bz, br, bs, out);
}
